// SpectralGatingNetwork_47897475285075
// MI455X (gfx1250) — matrix operand order repaired and checked by reading, not compiled, not run on hardware
//
#include <hip/hip_runtime.h>
#include <hip/hip_bf16.h>

// out[b,:] = M @ x[b,:] + bias, where M (512x512) and bias are built on-device
// from conv_w / conv_b / complex_weight (the whole reference pipeline is affine
// because the 0-width channel-attention MLP makes the gate a constant 0.5).

typedef __attribute__((ext_vector_type(16))) _Float16 v16h;
typedef __attribute__((ext_vector_type(8)))  float    v8f;

#define NROWS 16384
#define NK    512

// ---------------------------------------------------------------------------
// Kernel 1: per-channel circulant kernels K_c[di][dj]  (3 x 32 x 16 floats)
// K_c[di,dj] = (1/1024) * sum_{f=0..8} a_f sum_{k=0..31} wr*cos(phi) - wi*sin(phi)
// phi = 2*pi*(k*di/32 + f*dj/16), a_0=a_8=1, a_f=2 otherwise.
// (1/1024 = 0.5 gate / (sqrt(16)*sqrt(32)*sqrt(512)))
// ---------------------------------------------------------------------------
__global__ void sgn_build_k(const float* __restrict__ cw, float* __restrict__ Kc) {
    int t = blockIdx.x * blockDim.x + threadIdx.x;     // 0..1535
    if (t >= 1536) return;
    int c   = t >> 9;
    int rem = t & 511;
    int di  = rem >> 4;
    int dj  = rem & 15;
    const float twopi = 6.28318530717958647692f;
    float acc = 0.f;
    for (int f = 0; f <= 8; ++f) {
        float alpha = (f == 0 || f == 8) ? 1.f : 2.f;
        float fdj = twopi * (float)(f * dj) * (1.0f / 16.0f);
        for (int k = 0; k < 32; ++k) {
            float phi = twopi * (float)(k * di) * (1.0f / 32.0f) + fdj;
            float s = __sinf(phi);
            float co = __cosf(phi);
            float wr = cw[((k * 9 + f) * 3 + c) * 2 + 0];
            float wi = cw[((k * 9 + f) * 3 + c) * 2 + 1];
            acc += alpha * (wr * co - wi * s);
        }
    }
    Kc[c * 512 + di * 16 + dj] = acc * (1.0f / 1024.0f);
}

// ---------------------------------------------------------------------------
// Kernel 2: M[n',k] = (1/3) sum_{t=0..2} conv_w[t] * K_{c_t}[ (i'-i)%32, (j'-j)%16 ]
// with m = k + 512 t, p_t = m/3 (grid pos i*16+j), c_t = m%3.
// Written as f16, pre-swizzled into the WMMA B-operand lane layout:
//   tile (colTile ct, kChunk kc): lane = (kr>=16)*16 + col, half h = kr&15,
//   flat half index = ((ct*16+kc)*32 + lane)*16 + h.
// ---------------------------------------------------------------------------
__global__ void sgn_build_m(const float* __restrict__ conv_w,
                            const float* __restrict__ Kc,
                            _Float16* __restrict__ Bop) {
    int idx = blockIdx.x * blockDim.x + threadIdx.x;   // 0..262143
    int np = idx >> 9;          // output index n' (0..511)
    int k  = idx & 511;         // input index  k  (0..511)
    int ip = np >> 4, jp = np & 15;
    float acc = 0.f;
#pragma unroll
    for (int t = 0; t < 3; ++t) {
        int m   = k + 512 * t;
        int p   = m / 3;
        int cch = m - p * 3;
        int i = p >> 4, j = p & 15;
        int di = (ip - i) & 31;
        int dj = (jp - j) & 15;
        acc += conv_w[t] * Kc[cch * 512 + di * 16 + dj];
    }
    acc *= (1.f / 3.f);
    int ct = np >> 4, col = np & 15;
    int kc = k >> 5,  kr  = k & 31;
    int hh = kr >> 4, h   = kr & 15;
    int lane = hh * 16 + col;
    Bop[(((size_t)ct * 16 + kc) * 32 + lane) * 16 + h] = (_Float16)acc;
}

// ---------------------------------------------------------------------------
// Kernel 3: bias[n'] = (1/3) sum over all 1536 conv outputs of conv_b path
// ---------------------------------------------------------------------------
__global__ void sgn_build_bias(const float* __restrict__ cb,
                               const float* __restrict__ Kc,
                               float* __restrict__ bias) {
    int np = blockIdx.x * blockDim.x + threadIdx.x;    // 0..511
    if (np >= 512) return;
    int ip = np >> 4, jp = np & 15;
    float acc = 0.f;
    for (int m = 0; m < 1536; ++m) {
        int t   = m >> 9;
        int p   = m / 3;
        int cch = m - p * 3;
        int i = p >> 4, j = p & 15;
        int di = (ip - i) & 31;
        int dj = (jp - j) & 15;
        acc += cb[t] * Kc[cch * 512 + di * 16 + dj];
    }
    bias[np] = acc * (1.f / 3.f);
}

// ---------------------------------------------------------------------------
// Kernel 4: out = X(16384x512,f32) @ M^T + bias via v_wmma_f32_16x16x32_f16.
// Block = 256 threads = 8 waves; each wave owns one 16x16 output tile
// (block covers 128 rows x 16 cols). K loop: 16 chunks of 32.
// A operand: f32 loads of x converted to f16 in-register, placed in contiguous k order
// (lane<16: K=kk..kk+15; lane>=16: +16).
// B operand: one contiguous 32-byte load per lane from the pre-swizzled Bop.
// ---------------------------------------------------------------------------
__global__ __launch_bounds__(256) void sgn_gemm(const float* __restrict__ X,
                                                const _Float16* __restrict__ Bop,
                                                const float* __restrict__ bias,
                                                float* __restrict__ out) {
    const int lane    = threadIdx.x & 31;
    const int wave    = threadIdx.x >> 5;
    const int laneLow = lane & 15;
    const int hi      = lane >> 4;                 // 0 or 1
    const int rowBase = blockIdx.x * 128 + wave * 16;
    const int colTile = blockIdx.y;

    const float bv = bias[colTile * 16 + laneLow];
    v8f c;
#pragma unroll
    for (int g = 0; g < 8; ++g) c[g] = bv;

    const float*    xrow  = X + (size_t)(rowBase + laneLow) * NK;
    const _Float16* bbase = Bop + ((size_t)colTile * 16) * 512 + (size_t)lane * 16;

#pragma unroll 4
    for (int it = 0; it < 16; ++it) {
        const int kk   = it * 32;
        const int off0 = kk + hi * 16;
        float4 q0 = *(const float4*)(xrow + off0);
        float4 q1 = *(const float4*)(xrow + off0 + 4);
        float4 q2 = *(const float4*)(xrow + off0 + 8);
        float4 q3 = *(const float4*)(xrow + off0 + 12);
        v16h a;
        a[0]  = (_Float16)q0.x; a[1]  = (_Float16)q0.y;
        a[2]  = (_Float16)q0.z; a[3]  = (_Float16)q0.w;
        a[4]  = (_Float16)q1.x; a[5]  = (_Float16)q1.y;
        a[6]  = (_Float16)q1.z; a[7]  = (_Float16)q1.w;
        a[8]  = (_Float16)q2.x; a[9]  = (_Float16)q2.y;
        a[10] = (_Float16)q2.z; a[11] = (_Float16)q2.w;
        a[12] = (_Float16)q3.x; a[13] = (_Float16)q3.y;
        a[14] = (_Float16)q3.z; a[15] = (_Float16)q3.w;

        v16h b = *(const v16h*)(bbase + (size_t)it * 512);

        c = __builtin_amdgcn_wmma_f32_16x16x32_f16(
                /*neg_a=*/false, a, /*neg_b=*/false, b,
                /*c_mod=*/(short)0, c, /*reuse_a=*/false, /*reuse_b=*/false);
    }

    // C/D layout: VGPR g -> row = rowBase + g + 8*hi, col = colTile*16 + laneLow
    const int col = colTile * 16 + laneLow;
#pragma unroll
    for (int g = 0; g < 8; ++g) {
        out[(size_t)(rowBase + g + hi * 8) * NK + col] = c[g];
    }
}

extern "C" void kernel_launch(void* const* d_in, const int* in_sizes, int n_in,
                              void* d_out, int out_size, void* d_ws, size_t ws_size,
                              hipStream_t stream) {
    const float* x       = (const float*)d_in[0];   // (16384, 512, 1) f32
    const float* conv_w  = (const float*)d_in[1];   // (3,1,1)
    const float* conv_b  = (const float*)d_in[2];   // (3,)
    const float* cweight = (const float*)d_in[3];   // (32,9,3,2)
    float* out = (float*)d_out;                     // (16384, 512, 1) f32

    char* ws = (char*)d_ws;
    float*    Kc   = (float*)ws;                    // 1536 f32  (6 KB)
    float*    bias = (float*)(ws + 6144);           // 512  f32  (2 KB)
    _Float16* Bop  = (_Float16*)(ws + 8192);        // 262144 f16 (512 KB)

    sgn_build_k   <<<6, 256, 0, stream>>>(cweight, Kc);
    sgn_build_m   <<<1024, 256, 0, stream>>>(conv_w, Kc, Bop);
    sgn_build_bias<<<2, 256, 0, stream>>>(conv_b, Kc, bias);

    sgn_gemm<<<dim3(NROWS / 128, NK / 16), 256, 0, stream>>>(x, Bop, bias, out);
}
